// GCNForNode_47175920779583
// MI455X (gfx1250) — matrix operand order repaired and checked by reading, not compiled, not run on hardware
//
#include <hip/hip_runtime.h>

typedef __attribute__((ext_vector_type(16))) _Float16 v16h;
typedef __attribute__((ext_vector_type(8)))  float    v8f;

#define NFEAT 512
#define HID   64
#define NCLS  40
#define NCLSP 48   // padded to 3 WMMA n-tiles

// ---------------- threefry2x32-20, key = (0, 42)  (matches jax.random.key(42)) ----
__device__ __forceinline__ unsigned rotl32(unsigned v, int r) { return (v << r) | (v >> (32 - r)); }

__device__ __forceinline__ void threefry2x32_42(unsigned c0, unsigned c1,
                                                unsigned& o0, unsigned& o1) {
  const unsigned ks0 = 0u, ks1 = 42u, ks2 = 0u ^ 42u ^ 0x1BD11BDAu;
  unsigned x0 = c0 + ks0, x1 = c1 + ks1;
#define TF_R(r) { x0 += x1; x1 = rotl32(x1, r); x1 ^= x0; }
  TF_R(13) TF_R(15) TF_R(26) TF_R(6)   x0 += ks1; x1 += ks2 + 1u;
  TF_R(17) TF_R(29) TF_R(16) TF_R(24)  x0 += ks2; x1 += ks0 + 2u;
  TF_R(13) TF_R(15) TF_R(26) TF_R(6)   x0 += ks0; x1 += ks1 + 3u;
  TF_R(17) TF_R(29) TF_R(16) TF_R(24)  x0 += ks1; x1 += ks2 + 4u;
  TF_R(13) TF_R(15) TF_R(26) TF_R(6)   x0 += ks2; x1 += ks0 + 5u;
#undef TF_R
  o0 = x0; o1 = x1;
}

// ---------------- small elementwise kernels ---------------------------------------
__global__ void init_deg_kernel(float* deg, int n) {
  int i = blockIdx.x * blockDim.x + threadIdx.x;
  if (i < n) deg[i] = 1.0f;                      // self-loop contributes 1
}

__global__ void edge_deg_kernel(const int* __restrict__ dst, float* deg, int e) {
  int i = blockIdx.x * blockDim.x + threadIdx.x;
  if (i < e) unsafeAtomicAdd(&deg[dst[i]], 1.0f);
}

__global__ void rsqrt_deg_kernel(float* deg, int n) {
  int i = blockIdx.x * blockDim.x + threadIdx.x;
  if (i < n) deg[i] = rsqrtf(deg[i]);            // deg >= 1, max() not needed
}

__global__ void cvtW1_kernel(const float* __restrict__ W1, _Float16* __restrict__ W1T) {
  int o = blockIdx.x * blockDim.x + threadIdx.x;   // o = h*512 + k
  if (o < HID * NFEAT) {
    int h = o >> 9, k = o & (NFEAT - 1);
    W1T[o] = (_Float16)W1[k * HID + h];
  }
}

__global__ void cvtW2_kernel(const float* __restrict__ W2, _Float16* __restrict__ W2T) {
  int o = blockIdx.x * blockDim.x + threadIdx.x;   // o = r*64 + k  (r < 48)
  if (o < NCLSP * HID) {
    int r = o >> 6, k = o & (HID - 1);
    W2T[o] = (r < NCLS) ? (_Float16)W2[k * NCLS + r] : (_Float16)0.0f;
  }
}

// ---------------- GEMM1: h1[N,64] = f16wmma(x[N,512] @ W1[512,64]) ----------------
__global__ __launch_bounds__(256) void gemm1_wmma_kernel(
    const float* __restrict__ x, const _Float16* __restrict__ W1T,
    float* __restrict__ h1, int N) {
  __shared__ _Float16 lds[HID * NFEAT];            // 64 KB, W1 transposed [64][512] f16
  {
    int4* d = (int4*)lds;
    const int4* s = (const int4*)W1T;
    for (int i = threadIdx.x; i < (HID * NFEAT) / 8; i += 256) d[i] = s[i];
  }
  __syncthreads();

  const int lane = threadIdx.x & 31, wave = threadIdx.x >> 5;
  const int row0 = (blockIdx.x * 8 + wave) * 16;
  if (row0 >= N) return;                           // wave-uniform: EXEC stays all-1s
  const int m = lane & 15, hi = lane >> 4;

  const float* arow = x + (size_t)(row0 + m) * NFEAT;
  v8f c0 = {}, c1 = {}, c2 = {}, c3 = {};

  for (int k0 = 0; k0 < NFEAT; k0 += 32) {
    const float* p = arow + k0 + 16 * hi;
    float4 u0 = *(const float4*)(p);
    float4 u1 = *(const float4*)(p + 4);
    float4 u2 = *(const float4*)(p + 8);
    float4 u3 = *(const float4*)(p + 12);
    v16h a;
    a[0]  = (_Float16)u0.x; a[1]  = (_Float16)u0.y; a[2]  = (_Float16)u0.z; a[3]  = (_Float16)u0.w;
    a[4]  = (_Float16)u1.x; a[5]  = (_Float16)u1.y; a[6]  = (_Float16)u1.z; a[7]  = (_Float16)u1.w;
    a[8]  = (_Float16)u2.x; a[9]  = (_Float16)u2.y; a[10] = (_Float16)u2.z; a[11] = (_Float16)u2.w;
    a[12] = (_Float16)u3.x; a[13] = (_Float16)u3.y; a[14] = (_Float16)u3.z; a[15] = (_Float16)u3.w;

    const _Float16* bp = lds + (size_t)m * NFEAT + k0 + 16 * hi;
    v16h b0 = *(const v16h*)(bp);
    v16h b1 = *(const v16h*)(bp + 16 * NFEAT);
    v16h b2 = *(const v16h*)(bp + 32 * NFEAT);
    v16h b3 = *(const v16h*)(bp + 48 * NFEAT);

    c0 = __builtin_amdgcn_wmma_f32_16x16x32_f16(false, a, false, b0, (short)0, c0, false, false);
    c1 = __builtin_amdgcn_wmma_f32_16x16x32_f16(false, a, false, b1, (short)0, c1, false, false);
    c2 = __builtin_amdgcn_wmma_f32_16x16x32_f16(false, a, false, b2, (short)0, c2, false, false);
    c3 = __builtin_amdgcn_wmma_f32_16x16x32_f16(false, a, false, b3, (short)0, c3, false, false);
  }

  float* orow = h1 + (size_t)row0 * HID;
#pragma unroll
  for (int r = 0; r < 8; ++r) {
    int mr = r + 8 * hi;
    float* q = orow + (size_t)mr * HID + m;
    q[0] = c0[r]; q[16] = c1[r]; q[32] = c2[r]; q[48] = c3[r];
  }
}

// ---------------- layer-1 scatter: s1 = Â-weighted sum of h1 ----------------------
__global__ void selfinit1_kernel(const float* __restrict__ h1, const float* __restrict__ dinv,
                                 float* __restrict__ s1, int n) {   // n = N*64
  int t = blockIdx.x * blockDim.x + threadIdx.x;
  if (t < n) {
    int i = t >> 6;
    float di = dinv[i];
    s1[t] = h1[t] * di * di;                 // self-loop term also initializes buffer
  }
}

__global__ void edge_scatter1_kernel(const int* __restrict__ src, const int* __restrict__ dst,
                                     const float* __restrict__ dinv, const float* __restrict__ h1,
                                     float* s1, long long ef) {
  long long t = (long long)blockIdx.x * blockDim.x + threadIdx.x;
  if (t >= ef) return;
  int e = (int)(t >> 6), f = (int)(t & 63);
  int s = src[e], d = dst[e];
  float w = dinv[s] * dinv[d];
  unsafeAtomicAdd(&s1[(size_t)d * HID + f], h1[(size_t)s * HID + f] * w);
}

// ---------------- bias + relu + deterministic dropout (JAX threefry) --------------
__global__ void post1_kernel(float* s1, const float* __restrict__ b1, int H) { // H = N*64/2
  int j = blockIdx.x * blockDim.x + threadIdx.x;
  if (j >= H) return;
  unsigned r0, r1;
  threefry2x32_42((unsigned)j, (unsigned)(j + H), r0, r1);
  bool keep0 = (r0 >> 9) < (1u << 22);       // uniform[0,1) < 0.5
  bool keep1 = (r1 >> 9) < (1u << 22);
  float v0 = s1[j] + b1[j & 63];
  float v1 = s1[j + H] + b1[(j + H) & 63];
  v0 = v0 > 0.0f ? v0 : 0.0f;
  v1 = v1 > 0.0f ? v1 : 0.0f;
  s1[j]     = keep0 ? 2.0f * v0 : 0.0f;      // 1/(1-p) = 2
  s1[j + H] = keep1 ? 2.0f * v1 : 0.0f;
}

// ---------------- GEMM2: t2[N,40] = f16wmma(s1[N,64] @ W2[64,40]) -----------------
__global__ __launch_bounds__(256) void gemm2_wmma_kernel(
    const float* __restrict__ h2, const _Float16* __restrict__ W2T,
    float* __restrict__ t2, int N) {
  __shared__ _Float16 lds[NCLSP * HID];            // 6 KB, W2 transposed+padded [48][64] f16
  {
    int4* d = (int4*)lds;
    const int4* s = (const int4*)W2T;
    for (int i = threadIdx.x; i < (NCLSP * HID) / 8; i += 256) d[i] = s[i];
  }
  __syncthreads();

  const int lane = threadIdx.x & 31, wave = threadIdx.x >> 5;
  const int row0 = (blockIdx.x * 8 + wave) * 16;
  if (row0 >= N) return;
  const int m = lane & 15, hi = lane >> 4;

  const float* arow = h2 + (size_t)(row0 + m) * HID;
  v8f c0 = {}, c1 = {}, c2 = {};

#pragma unroll
  for (int k0 = 0; k0 < HID; k0 += 32) {
    const float* p = arow + k0 + 16 * hi;
    float4 u0 = *(const float4*)(p);
    float4 u1 = *(const float4*)(p + 4);
    float4 u2 = *(const float4*)(p + 8);
    float4 u3 = *(const float4*)(p + 12);
    v16h a;
    a[0]  = (_Float16)u0.x; a[1]  = (_Float16)u0.y; a[2]  = (_Float16)u0.z; a[3]  = (_Float16)u0.w;
    a[4]  = (_Float16)u1.x; a[5]  = (_Float16)u1.y; a[6]  = (_Float16)u1.z; a[7]  = (_Float16)u1.w;
    a[8]  = (_Float16)u2.x; a[9]  = (_Float16)u2.y; a[10] = (_Float16)u2.z; a[11] = (_Float16)u2.w;
    a[12] = (_Float16)u3.x; a[13] = (_Float16)u3.y; a[14] = (_Float16)u3.z; a[15] = (_Float16)u3.w;

    const _Float16* bp = lds + (size_t)m * HID + k0 + 16 * hi;
    v16h b0 = *(const v16h*)(bp);
    v16h b1 = *(const v16h*)(bp + 16 * HID);
    v16h b2 = *(const v16h*)(bp + 32 * HID);

    c0 = __builtin_amdgcn_wmma_f32_16x16x32_f16(false, a, false, b0, (short)0, c0, false, false);
    c1 = __builtin_amdgcn_wmma_f32_16x16x32_f16(false, a, false, b1, (short)0, c1, false, false);
    c2 = __builtin_amdgcn_wmma_f32_16x16x32_f16(false, a, false, b2, (short)0, c2, false, false);
  }

#pragma unroll
  for (int r = 0; r < 8; ++r) {
    int mr = r + 8 * hi;
    float* q = t2 + (size_t)(row0 + mr) * NCLS;
    q[m] = c0[r];
    q[16 + m] = c1[r];
    if (m < 8) q[32 + m] = c2[r];              // cols 40..47 are padding, dropped
  }
}

// ---------------- layer-2 scatter into d_out --------------------------------------
__global__ void selfinit2_kernel(const float* __restrict__ t2, const float* __restrict__ dinv,
                                 const float* __restrict__ b2, float* __restrict__ out, int n) {
  int t = blockIdx.x * blockDim.x + threadIdx.x;   // n = N*40
  if (t < n) {
    int i = t / NCLS, f = t - i * NCLS;
    float di = dinv[i];
    out[t] = t2[t] * di * di + b2[f];
  }
}

__global__ void edge_scatter2_kernel(const int* __restrict__ src, const int* __restrict__ dst,
                                     const float* __restrict__ dinv, const float* __restrict__ t2,
                                     float* out, long long ef) {
  long long t = (long long)blockIdx.x * blockDim.x + threadIdx.x;
  if (t >= ef) return;
  long long e = t / NCLS;
  int f = (int)(t - e * NCLS);
  int s = src[e], d = dst[e];
  float w = dinv[s] * dinv[d];
  unsafeAtomicAdd(&out[(size_t)d * NCLS + f], t2[(size_t)s * NCLS + f] * w);
}

// ---------------- launcher --------------------------------------------------------
static inline unsigned cdivll(long long a, long long b) { return (unsigned)((a + b - 1) / b); }

extern "C" void kernel_launch(void* const* d_in, const int* in_sizes, int n_in,
                              void* d_out, int out_size, void* d_ws, size_t ws_size,
                              hipStream_t stream) {
  (void)n_in; (void)out_size; (void)ws_size;
  const float* x  = (const float*)d_in[0];
  const int*   ei = (const int*)d_in[1];
  const float* W1 = (const float*)d_in[2];
  const float* b1 = (const float*)d_in[3];
  const float* W2 = (const float*)d_in[4];
  const float* b2 = (const float*)d_in[5];
  float* out = (float*)d_out;

  const int N = in_sizes[0] / NFEAT;               // 100000
  const int E = in_sizes[1] / 2;                   // 3200000
  const int* src = ei;
  const int* dst = ei + E;

  char* ws = (char*)d_ws;
  size_t off = 0;
  auto take = [&](size_t bytes) { size_t o = off; off += (bytes + 255) & ~(size_t)255; return o; };
  float*    deg = (float*)(ws + take((size_t)N * 4));                 // becomes dinv in place
  _Float16* W1T = (_Float16*)(ws + take((size_t)NFEAT * HID * 2));
  _Float16* W2T = (_Float16*)(ws + take((size_t)NCLSP * HID * 2));
  float*    h1  = (float*)(ws + take((size_t)N * HID * 4));
  float*    s1  = (float*)(ws + take((size_t)N * HID * 4));
  float*    t2  = h1;   // h1 is dead after edge_scatter1; reuse for t2 (N*40 < N*64)

  const int B = 256;
  init_deg_kernel<<<cdivll(N, B), B, 0, stream>>>(deg, N);
  edge_deg_kernel<<<cdivll(E, B), B, 0, stream>>>(dst, deg, E);
  rsqrt_deg_kernel<<<cdivll(N, B), B, 0, stream>>>(deg, N);
  cvtW1_kernel<<<cdivll(HID * NFEAT, B), B, 0, stream>>>(W1, W1T);
  cvtW2_kernel<<<cdivll(NCLSP * HID, B), B, 0, stream>>>(W2, W2T);

  gemm1_wmma_kernel<<<cdivll(N, 128), B, 0, stream>>>(x, W1T, h1, N);
  selfinit1_kernel<<<cdivll((long long)N * HID, B), B, 0, stream>>>(h1, deg, s1, N * HID);
  edge_scatter1_kernel<<<cdivll((long long)E * HID, B), B, 0, stream>>>(src, dst, deg, h1, s1,
                                                                        (long long)E * HID);
  post1_kernel<<<cdivll((long long)N * HID / 2, B), B, 0, stream>>>(s1, b1, N * HID / 2);

  gemm2_wmma_kernel<<<cdivll(N, 128), B, 0, stream>>>(s1, W2T, t2, N);
  selfinit2_kernel<<<cdivll((long long)N * NCLS, B), B, 0, stream>>>(t2, deg, b2, out, N * NCLS);
  edge_scatter2_kernel<<<cdivll((long long)E * NCLS, B), B, 0, stream>>>(src, dst, deg, t2, out,
                                                                         (long long)E * NCLS);
}
